// LSTM_1_37735582663296
// MI455X (gfx1250) — compile-verified, matrix operand order checked by reading, not run on hardware
//
#include <hip/hip_runtime.h>
#include <hip/hip_bf16.h>

// LSTM: B=32, T=2048, D=128, U=128 (4U = 512 packed gates: i,f,g,o)
//
// Kernel 1 (parallel, memory-bound):  XW = X@W_x + bias, one block per t,
//   A rows = batch. C tiles stored in WMMA C-fragment layout
//   xw[((t*2+mb)*32+nt)*256 + lane*8 + r]  so kernel 2 loads them with b128s.
// Kernel 2 (serial chain, 2 persistent WGs of 8 waves): wave w owns units
//   [16w,16w+16) == N-tiles {w, w+8, w+16, w+24} (i,f,g,o)  ->  all four gate
//   accumulators + cell state c live in that wave's registers. W_h fragments
//   resident in VGPRs; h ping-pongs through 8KB of LDS; xW prefetched 1 step
//   ahead; ONE barrier per timestep.

#define T_ 2048
#define B_ 32
#define D_ 128
#define NU 512  // 4*U

typedef __attribute__((ext_vector_type(16))) __bf16 v16bf;
typedef __attribute__((ext_vector_type(8)))  float  v8f;

__device__ __forceinline__ float sigmoid_f(float x) {
  return 1.0f / (1.0f + __expf(-x));
}
__device__ __forceinline__ float tanh_f(float x) {
  float e = __expf(2.0f * x);
  return (e - 1.0f) / (e + 1.0f);
}

// 16-bit A/B fragment K-mapping inside a 32-wide K-step, lane-half g:
//   element e  ->  K = (e&7) + 16*(e>>3) + 8*g   (two K-contiguous runs of 8)

// ---------------------------------------------------------------------------
// Kernel 1: one block per t, 512 threads (16 waves).
// wave -> (mb = wave>>3, nt0 = wave&7), computes tiles nt0+8q, q=0..3.
// ---------------------------------------------------------------------------
__global__ __launch_bounds__(512) void lstm_xw_gemm(
    const float* __restrict__ x,     // [B][T][D] f32
    const float* __restrict__ Wx,    // [D][4U]   f32
    const float* __restrict__ bias,  // [4U]      f32
    float* __restrict__ xw)          // [T][2][32][32][8] f32 tile layout
{
  __shared__ __align__(16) float xs[32][D_];  // 16 KB: x_t for all 32 batch rows

  const int t   = blockIdx.x;
  const int tid = threadIdx.x;

  // Cooperative coalesced load of x[:, t, :] (rows strided by T*D).
  {
    const int row = tid >> 4;          // 0..31
    const int off = (tid & 15) * 8;    // 0..120
    const float* src = x + (size_t)row * T_ * D_ + (size_t)t * D_ + off;
#pragma unroll
    for (int e = 0; e < 8; ++e) xs[row][off + e] = src[e];
  }
  __syncthreads();

  const int wave = tid >> 5;
  const int lane = tid & 31;
  const int g    = lane >> 4;
  const int ln   = lane & 15;
  const int mb   = wave >> 3;   // batch half (M tile)
  const int nt0  = wave & 7;

  // A fragments: rows = batch mb*16+ln, K over D.
  v16bf a[4];
#pragma unroll
  for (int ks = 0; ks < 4; ++ks) {
    const int kb = ks * 32 + 8 * g;
#pragma unroll
    for (int e = 0; e < 8; ++e) {
      a[ks][e]     = (__bf16)xs[mb * 16 + ln][kb + e];
      a[ks][e + 8] = (__bf16)xs[mb * 16 + ln][kb + 16 + e];
    }
  }

#pragma unroll
  for (int q = 0; q < 4; ++q) {
    const int nt = nt0 + 8 * q;
    const int n  = nt * 16 + ln;

    v16bf bwq[4];
#pragma unroll
    for (int ks = 0; ks < 4; ++ks) {
      const int kb = ks * 32 + 8 * g;
#pragma unroll
      for (int e = 0; e < 8; ++e) {
        bwq[ks][e]     = (__bf16)Wx[(size_t)(kb + e) * NU + n];
        bwq[ks][e + 8] = (__bf16)Wx[(size_t)(kb + 16 + e) * NU + n];
      }
    }

    const float bv = bias[n];
    v8f acc;
#pragma unroll
    for (int r = 0; r < 8; ++r) acc[r] = bv;

#pragma unroll
    for (int ks = 0; ks < 4; ++ks)
      acc = __builtin_amdgcn_wmma_f32_16x16x32_bf16(
          false, a[ks], false, bwq[ks], (short)0, acc, false, false);

    // Store in C-fragment layout: 8 consecutive floats per lane (-> b128 x2).
    float* dst = xw + (((size_t)t * 2 + mb) * 32 + nt) * 256 + lane * 8;
#pragma unroll
    for (int r = 0; r < 8; ++r) dst[r] = acc[r];
  }
}

// ---------------------------------------------------------------------------
// Kernel 2: recurrence. 2 blocks x 256 threads (8 waves), block = batch half.
// ---------------------------------------------------------------------------
__global__ __launch_bounds__(256) void lstm_recurrent(
    const float* __restrict__ xw,   // tile-layout preactivations
    const float* __restrict__ Wh,   // [U][4U] f32
    float* __restrict__ out)        // [T][B][U] f32
{
  __shared__ __align__(16) __bf16 hbuf[2][16][128];  // ping-pong h, 8 KB

  const int tid  = threadIdx.x;
  const int wave = tid >> 5;       // 0..7  (unit group)
  const int lane = tid & 31;
  const int g    = lane >> 4;
  const int ln   = lane & 15;
  const int mb   = blockIdx.x;
  const int b0   = mb * 16;
  const int u    = wave * 16 + ln; // this lane's hidden unit

  // h(-1) = 0 in buffer 0.
  for (int i = tid; i < 16 * 128; i += 256)
    hbuf[0][i >> 7][i & 127] = (__bf16)0.0f;

  // Resident W_h fragments: gate q uses N-tile q*8+wave (cols q*128+u).
  v16bf bw[4][4];
#pragma unroll
  for (int q = 0; q < 4; ++q) {
    const int n = (q * 8 + wave) * 16 + ln;
#pragma unroll
    for (int ks = 0; ks < 4; ++ks) {
      const int kb = ks * 32 + 8 * g;
#pragma unroll
      for (int e = 0; e < 8; ++e) {
        bw[q][ks][e]     = (__bf16)Wh[(size_t)(kb + e) * NU + n];
        bw[q][ks][e + 8] = (__bf16)Wh[(size_t)(kb + 16 + e) * NU + n];
      }
    }
  }

  // Cell state in registers: c[r] is row m = r+8g, unit u.
  float c[8];
#pragma unroll
  for (int r = 0; r < 8; ++r) c[r] = 0.0f;

  // Prefetch xW tiles for t = 0 (2x b128 per gate).
  v8f nxt[4];
#pragma unroll
  for (int q = 0; q < 4; ++q) {
    const float* p =
        xw + (((size_t)0 * 2 + mb) * 32 + (q * 8 + wave)) * 256 + lane * 8;
#pragma unroll
    for (int r = 0; r < 8; ++r) nxt[q][r] = p[r];
  }
  __syncthreads();

  for (int t = 0; t < T_; ++t) {
    const int pp = t & 1;

    // A fragments = h(t-1) from LDS (two 16B reads per K-step).
    v16bf a[4];
#pragma unroll
    for (int ks = 0; ks < 4; ++ks) {
      union { uint4 u4[2]; v16bf v; } au;
      au.u4[0] = *reinterpret_cast<const uint4*>(&hbuf[pp][ln][ks * 32 + 8 * g]);
      au.u4[1] = *reinterpret_cast<const uint4*>(&hbuf[pp][ln][ks * 32 + 16 + 8 * g]);
      a[ks] = au.v;
    }

    // Consume prefetched xW, immediately issue loads for t+1 (latency hides
    // behind this step's WMMA + gate math).
    v8f acc[4];
#pragma unroll
    for (int q = 0; q < 4; ++q) acc[q] = nxt[q];
    const int tn = (t + 1 < T_) ? (t + 1) : t;
#pragma unroll
    for (int q = 0; q < 4; ++q) {
      const float* p =
          xw + (((size_t)tn * 2 + mb) * 32 + (q * 8 + wave)) * 256 + lane * 8;
#pragma unroll
      for (int r = 0; r < 8; ++r) nxt[q][r] = p[r];
    }

    // z[q] = h(t-1) @ W_h  (+ precomputed x@W_x + bias): 16 WMMAs,
    // 4 independent accumulator chains.
#pragma unroll
    for (int q = 0; q < 4; ++q) {
#pragma unroll
      for (int ks = 0; ks < 4; ++ks)
        acc[q] = __builtin_amdgcn_wmma_f32_16x16x32_bf16(
            false, a[ks], false, bw[q][ks], (short)0, acc[q], false, false);
    }

    // Gates + state entirely in registers (lane-aligned across the 4 accs).
#pragma unroll
    for (int r = 0; r < 8; ++r) {
      const int m  = r + 8 * g;
      const float ig = sigmoid_f(acc[0][r]);
      const float fg = sigmoid_f(acc[1][r]);
      const float gg = tanh_f(acc[2][r]);
      const float og = sigmoid_f(acc[3][r]);
      c[r] = ig * gg + fg * c[r];
      const float h = og * tanh_f(c[r]);
      hbuf[pp ^ 1][m][u] = (__bf16)h;                     // next step's A source
      out[(size_t)t * B_ * 128 + (size_t)(b0 + m) * 128 + u] = h;  // result
    }

    __syncthreads();  // h(t) visible before anyone reads it at t+1
  }
}

// ---------------------------------------------------------------------------
extern "C" void kernel_launch(void* const* d_in, const int* in_sizes, int n_in,
                              void* d_out, int out_size, void* d_ws, size_t ws_size,
                              hipStream_t stream) {
  const float* x    = (const float*)d_in[0];  // [B,T,D]
  const float* Wx   = (const float*)d_in[1];  // [D,4U]
  const float* Wh   = (const float*)d_in[2];  // [U,4U]
  const float* bias = (const float*)d_in[3];  // [4U]
  float* out = (float*)d_out;                 // [T,B,U] f32
  float* xw  = (float*)d_ws;                  // T*2*32*256*4 = 128 MB scratch

  lstm_xw_gemm<<<dim3(T_), 512, 0, stream>>>(x, Wx, bias, xw);
  lstm_recurrent<<<dim3(2), 256, 0, stream>>>(xw, Wh, out);
}
